// TransformerPointerNet2_90881507983857
// MI455X (gfx1250) — matrix operand order repaired and checked by reading, not compiled, not run on hardware
//
#include <hip/hip_runtime.h>
#include <hip/hip_bf16.h>
#include <math.h>

// ---------------------------------------------------------------------------
// CDNA5 (gfx1250) implementation of the transformer pointer network.
// - All large GEMMs use v_wmma_f32_16x16x32_f16 (wave32, 16x16x32 fragments).
// - Decoder K/V projections + pointer keys hoisted out of the 100-step loop.
// - Attention / softmax / sampling: one wave per row, LDS + shfl reductions.
// ---------------------------------------------------------------------------

typedef __attribute__((ext_vector_type(16))) _Float16 v16h;
typedef __attribute__((ext_vector_type(8)))  _Float16 v8h;
typedef __attribute__((ext_vector_type(8)))  float    v8f;

constexpr int H_    = 256;
constexpr int NH_   = 8;
constexpr int DH_   = 32;     // head dim == wave32 width
constexpr int DFF_  = 1024;
constexpr int NL_   = 3;
constexpr int IN_   = 16;
constexpr int SEQ_  = 100;
constexpr int B_    = 256;
constexpr int NS_   = 8;
constexpr int BM_   = B_ * NS_;        // 2048
constexpr int ROWS_ = SEQ_ * B_;       // 25600
constexpr float NEG_  = -1e9f;
constexpr float CLIP_ = 10.0f;

// ---------------------------------------------------------------------------
// wave32 reductions
// ---------------------------------------------------------------------------
__device__ __forceinline__ float wave_max(float v) {
#pragma unroll
  for (int o = 16; o > 0; o >>= 1) v = fmaxf(v, __shfl_xor(v, o, 32));
  return v;
}
__device__ __forceinline__ float wave_sum(float v) {
#pragma unroll
  for (int o = 16; o > 0; o >>= 1) v += __shfl_xor(v, o, 32);
  return v;
}

// ---------------------------------------------------------------------------
// WMMA fragment loaders (layouts per CDNA5 ISA §7.12.2)
// A (16x32 f16, row-major [M,K]): lanes 0-15 -> M, K {0..7,16..23};
//                                 lanes 16-31 -> M, K {8..15,24..31}
// B = W^T where W is row-major [N,K]: lane n holds K in the same order as A;
//   lanes 0-15 -> K {0..7,16..23}, lanes 16-31 -> K {8..15,24..31}.
// ---------------------------------------------------------------------------
__device__ __forceinline__ v16h load_a_frag(const _Float16* __restrict__ A,
                                            int lda, int m0, int k0, int lane) {
  const int m  = m0 + (lane & 15);
  const int kb = k0 + ((lane >> 4) << 3);
  const _Float16* p = A + (size_t)m * lda + kb;
  v8h lo = *(const v8h*)(p);
  v8h hi = *(const v8h*)(p + 16);
  v16h r;
#pragma unroll
  for (int i = 0; i < 8; ++i) { r[i] = lo[i]; r[i + 8] = hi[i]; }
  return r;
}

__device__ __forceinline__ v16h load_b_frag(const _Float16* __restrict__ W,
                                            int ldw, int n0, int k0, int lane) {
  const int n  = n0 + (lane & 15);
  const int kb = k0 + ((lane >> 4) << 3);
  const _Float16* p = W + (size_t)n * ldw + kb;
  v8h lo = *(const v8h*)(p);
  v8h hi = *(const v8h*)(p + 16);
  v16h r;
#pragma unroll
  for (int i = 0; i < 8; ++i) { r[i] = lo[i]; r[i + 8] = hi[i]; }
  return r;
}

// ---------------------------------------------------------------------------
// GEMM: C[M,N] = act(A[M,K] * W[N,K]^T + bias)
// Block = 128 threads = 4 waves; each wave computes a 32x32 tile (2x2 WMMA).
// flags: bit0 = add bias, bit1 = relu.  Cf (f32) and/or Ch (f16) outputs.
// M,N must be multiples of 64; K a multiple of 32 (true for all call sites).
// ---------------------------------------------------------------------------
__global__ void __launch_bounds__(128)
gemm_wmma_kernel(const _Float16* __restrict__ A, const _Float16* __restrict__ W,
                 const float* __restrict__ bias, float* __restrict__ Cf,
                 _Float16* __restrict__ Ch, int M, int N, int K, int flags) {
  const int lane = threadIdx.x & 31;
  const int wave = threadIdx.x >> 5;
  const int nBlk = blockIdx.x * 64 + (wave & 1) * 32;
  const int mBlk = blockIdx.y * 64 + (wave >> 1) * 32;

  v8f acc00 = {}, acc01 = {}, acc10 = {}, acc11 = {};

  for (int k = 0; k < K; k += 32) {
    if (k + 64 <= K) {  // warm L2/L0 for the next k-tile (global_prefetch_b8)
      __builtin_prefetch(A + (size_t)(mBlk + (lane & 15)) * K + k + 32, 0, 1);
      __builtin_prefetch(W + (size_t)(nBlk + (lane & 15)) * K + k + 32, 0, 1);
    }
    v16h a0 = load_a_frag(A, K, mBlk,      k, lane);
    v16h a1 = load_a_frag(A, K, mBlk + 16, k, lane);
    v16h b0 = load_b_frag(W, K, nBlk,      k, lane);
    v16h b1 = load_b_frag(W, K, nBlk + 16, k, lane);
    acc00 = __builtin_amdgcn_wmma_f32_16x16x32_f16(false, a0, false, b0, (short)0, acc00, false, false);
    acc01 = __builtin_amdgcn_wmma_f32_16x16x32_f16(false, a0, false, b1, (short)0, acc01, false, false);
    acc10 = __builtin_amdgcn_wmma_f32_16x16x32_f16(false, a1, false, b0, (short)0, acc10, false, false);
    acc11 = __builtin_amdgcn_wmma_f32_16x16x32_f16(false, a1, false, b1, (short)0, acc11, false, false);
  }

  // Epilogue: C/D layout -> VGPR r holds M = r + 8*(lane>=16), N = lane&15.
  const int nCol0 = nBlk + (lane & 15);
  const int mOff  = ((lane >> 4) << 3);
#pragma unroll
  for (int ni = 0; ni < 2; ++ni) {
    const int n = nCol0 + ni * 16;
    const float bv = (flags & 1) ? bias[n] : 0.0f;
#pragma unroll
    for (int mi = 0; mi < 2; ++mi) {
      v8f c = (mi == 0) ? (ni == 0 ? acc00 : acc01) : (ni == 0 ? acc10 : acc11);
#pragma unroll
      for (int r = 0; r < 8; ++r) {
        const int m = mBlk + mi * 16 + r + mOff;
        float v = c[r] + bv;
        if (flags & 2) v = fmaxf(v, 0.0f);
        const size_t idx = (size_t)m * N + n;
        if (Cf) Cf[idx] = v;
        if (Ch) Ch[idx] = (_Float16)v;
      }
    }
  }
}

// ---------------------------------------------------------------------------
// Input projection: x = input @ proj_w^T + proj_b   (K=16, tiny -> VALU)
// ---------------------------------------------------------------------------
__global__ void in_proj_kernel(const float* __restrict__ in,
                               const float* __restrict__ w,
                               const float* __restrict__ bias,
                               float* __restrict__ x, _Float16* __restrict__ xh) {
  const int idx = blockIdx.x * blockDim.x + threadIdx.x;
  if (idx >= ROWS_ * H_) return;
  const int r = idx / H_, n = idx % H_;
  float a = bias[n];
#pragma unroll
  for (int k = 0; k < IN_; ++k) a = fmaf(in[(size_t)r * IN_ + k], w[n * IN_ + k], a);
  x[idx] = a;
  xh[idx] = (_Float16)a;
}

// ---------------------------------------------------------------------------
// Residual + LayerNorm (one wave per row of 256)
// ---------------------------------------------------------------------------
__global__ void __launch_bounds__(128)
ln_kernel(float* __restrict__ x, const float* __restrict__ y,
          const float* __restrict__ g, const float* __restrict__ beta,
          _Float16* __restrict__ xh, int rows) {
  const int lane = threadIdx.x & 31, w4 = threadIdx.x >> 5;
  const int row = blockIdx.x * 4 + w4;
  if (row >= rows) return;
  float v[8]; float s = 0.0f;
#pragma unroll
  for (int j = 0; j < 8; ++j) {
    const size_t idx = (size_t)row * H_ + j * 32 + lane;
    v[j] = x[idx] + y[idx];
    s += v[j];
  }
  const float mu = wave_sum(s) * (1.0f / H_);
  float q = 0.0f;
#pragma unroll
  for (int j = 0; j < 8; ++j) { const float d = v[j] - mu; q = fmaf(d, d, q); }
  const float rs = rsqrtf(wave_sum(q) * (1.0f / H_) + 1e-5f);
#pragma unroll
  for (int j = 0; j < 8; ++j) {
    const int e = j * 32 + lane;
    const float o = (v[j] - mu) * rs * g[e] + beta[e];
    const size_t idx = (size_t)row * H_ + e;
    x[idx] = o;
    xh[idx] = (_Float16)o;
  }
}

// ---------------------------------------------------------------------------
// Encoder self-attention: one wave per (b, h, q) row.  dh = 32 = lanes.
// qkv layout: row (l*B + b), cols [q(0..255) | k(256..511) | v(512..767)]
// ---------------------------------------------------------------------------
__global__ void __launch_bounds__(128)
enc_attn_kernel(const float* __restrict__ qkv, const float* __restrict__ mask,
                _Float16* __restrict__ outh) {
  const int lane = threadIdx.x & 31, w4 = threadIdx.x >> 5;
  const int wid = blockIdx.x * 4 + w4;                 // B_*NH_*SEQ_ waves
  const int q = wid % SEQ_;
  const int bh = wid / SEQ_;
  const int h = bh & (NH_ - 1);
  const int b = bh / NH_;
  __shared__ float sQ[4][DH_];
  __shared__ float sA[4][SEQ_];
  sQ[w4][lane] = qkv[(size_t)(q * B_ + b) * (3 * H_) + h * DH_ + lane];
  __syncthreads();

  const float scale = 0.17677669529663687f;  // 1/sqrt(32)
  float sc[4];
#pragma unroll
  for (int j = 0; j < 4; ++j) {
    const int kk = lane + j * 32;
    float s = NEG_;
    if (kk < SEQ_ && mask[b * SEQ_ + kk] != 0.0f) {
      const float* kr = qkv + (size_t)(kk * B_ + b) * (3 * H_) + H_ + h * DH_;
      float d = 0.0f;
#pragma unroll
      for (int e = 0; e < DH_; ++e) d = fmaf(sQ[w4][e], kr[e], d);
      s = d * scale;
    }
    sc[j] = s;
  }
  float m = fmaxf(fmaxf(sc[0], sc[1]), fmaxf(sc[2], sc[3]));
  m = wave_max(m);
  float ex[4]; float se = 0.0f;
#pragma unroll
  for (int j = 0; j < 4; ++j) {
    ex[j] = (sc[j] > 0.5f * NEG_) ? __expf(sc[j] - m) : 0.0f;
    se += ex[j];
  }
  se = wave_sum(se);
  const float inv = 1.0f / se;
#pragma unroll
  for (int j = 0; j < 4; ++j) {
    const int kk = lane + j * 32;
    if (kk < SEQ_) sA[w4][kk] = ex[j] * inv;
  }
  __syncthreads();
  float o = 0.0f;
  for (int kk = 0; kk < SEQ_; ++kk)
    o = fmaf(sA[w4][kk], qkv[(size_t)(kk * B_ + b) * (3 * H_) + 2 * H_ + h * DH_ + lane], o);
  outh[(size_t)(q * B_ + b) * H_ + h * DH_ + lane] = (_Float16)o;
}

// ---------------------------------------------------------------------------
// Graph embedding: mean over sequence dim
// ---------------------------------------------------------------------------
__global__ void graph_mean_kernel(const float* __restrict__ x, float* __restrict__ graph) {
  const int idx = blockIdx.x * blockDim.x + threadIdx.x;
  if (idx >= B_ * H_) return;
  const int b = idx / H_, e = idx % H_;
  float s = 0.0f;
  for (int l = 0; l < SEQ_; ++l) s += x[(size_t)(l * B_ + b) * H_ + e];
  graph[idx] = s * (1.0f / SEQ_);
}

// ---------------------------------------------------------------------------
// Decoder state init (mask_dec, pm, lps, prev_idx) -- ws is poisoned, so init.
// ---------------------------------------------------------------------------
__global__ void dec_init_kernel(const float* __restrict__ maskB,
                                float* __restrict__ mask_dec, int* __restrict__ pm,
                                float* __restrict__ lps, int* __restrict__ prev_idx) {
  const int idx = blockIdx.x * blockDim.x + threadIdx.x;
  if (idx < BM_ * SEQ_) {
    const int bm = idx / SEQ_, l = idx % SEQ_;
    mask_dec[idx] = maskB[(bm / NS_) * SEQ_ + l];
    pm[idx] = 0;
  }
  if (idx < BM_) { lps[idx] = 0.0f; prev_idx[idx] = 0; }
}

// ---------------------------------------------------------------------------
// Decoder context build: ctx[bm] = [graph | prev | adjacent]  -> f16 for GEMM
// ---------------------------------------------------------------------------
__global__ void ctx_build_kernel(const float* __restrict__ x,
                                 const float* __restrict__ graph,
                                 const float* __restrict__ v1,
                                 const float* __restrict__ v2,
                                 const int* __restrict__ pm,
                                 const int* __restrict__ prev_idx,
                                 _Float16* __restrict__ ctxh, int t) {
  const int idx = blockIdx.x * blockDim.x + threadIdx.x;
  if (idx >= BM_ * 3 * H_) return;
  const int bm = idx / (3 * H_);
  const int c = idx % (3 * H_);
  const int b = bm / NS_;
  float v;
  if (c < H_) {
    v = graph[b * H_ + c];
  } else if (c < 2 * H_) {
    const int e = c - H_;
    v = (t == 0) ? v2[e] : x[(size_t)(prev_idx[bm] * B_ + b) * H_ + e];
  } else {
    const int e = c - 2 * H_;
    if (t <= 1) {
      v = v1[e];
    } else {
      const int at = (t >= 10) ? (t - 10) : ((t - 1 > 0) ? t - 1 : 0);  // NoC mesh adjacency, w=10
      int ai = pm[bm * SEQ_ + at];
      if (ai < 0) ai = 0;
      v = x[(size_t)(ai * B_ + b) * H_ + e];
    }
  }
  ctxh[(size_t)bm * 3 * H_ + c] = (_Float16)v;
}

// ---------------------------------------------------------------------------
// Decoder cross-attention (Lq=1): one wave per (bm, h); K/V precomputed per b.
// ---------------------------------------------------------------------------
__global__ void __launch_bounds__(128)
dec_attn_kernel(const float* __restrict__ qdec, const float* __restrict__ Kd,
                const float* __restrict__ Vd, const float* __restrict__ mask_dec,
                _Float16* __restrict__ glh) {
  const int lane = threadIdx.x & 31, w4 = threadIdx.x >> 5;
  const int wid = blockIdx.x * 4 + w4;                 // BM_*NH_ waves
  const int h = wid & (NH_ - 1);
  const int bm = wid >> 3;
  const int b = bm / NS_;
  __shared__ float sQ[4][DH_];
  __shared__ float sA[4][SEQ_];
  sQ[w4][lane] = qdec[(size_t)bm * H_ + h * DH_ + lane];
  __syncthreads();

  const float scale = 0.17677669529663687f;
  float sc[4];
#pragma unroll
  for (int j = 0; j < 4; ++j) {
    const int kk = lane + j * 32;
    float s = NEG_;
    if (kk < SEQ_ && mask_dec[bm * SEQ_ + kk] != 0.0f) {
      const float* kr = Kd + (size_t)(kk * B_ + b) * H_ + h * DH_;
      float d = 0.0f;
#pragma unroll
      for (int e = 0; e < DH_; ++e) d = fmaf(sQ[w4][e], kr[e], d);
      s = d * scale;
    }
    sc[j] = s;
  }
  float m = fmaxf(fmaxf(sc[0], sc[1]), fmaxf(sc[2], sc[3]));
  m = wave_max(m);
  float ex[4]; float se = 0.0f;
#pragma unroll
  for (int j = 0; j < 4; ++j) {
    ex[j] = (sc[j] > 0.5f * NEG_) ? __expf(sc[j] - m) : 0.0f;
    se += ex[j];
  }
  se = wave_sum(se);
  const float inv = 1.0f / se;
#pragma unroll
  for (int j = 0; j < 4; ++j) {
    const int kk = lane + j * 32;
    if (kk < SEQ_) sA[w4][kk] = ex[j] * inv;
  }
  __syncthreads();
  float o = 0.0f;
  for (int kk = 0; kk < SEQ_; ++kk)
    o = fmaf(sA[w4][kk], Vd[(size_t)(kk * B_ + b) * H_ + h * DH_ + lane], o);
  glh[(size_t)bm * H_ + h * DH_ + lane] = (_Float16)o;
}

// ---------------------------------------------------------------------------
// Pointer logits + log-softmax + categorical sample + state update.
// One wave per bm.  (PRNG: deterministic hash of (key, t, bm).)
// ---------------------------------------------------------------------------
__device__ __forceinline__ float u01_hash(unsigned long long x) {
  x += 0x9E3779B97f4A7C15ULL;
  x = (x ^ (x >> 30)) * 0xBF58476D1CE4E5B9ULL;
  x = (x ^ (x >> 27)) * 0x94D049BB133111EBULL;
  x ^= x >> 31;
  return (float)(x >> 40) * (1.0f / 16777216.0f);
}

__global__ void __launch_bounds__(128)
ptr_sample_kernel(const float* __restrict__ qp, const float* __restrict__ kp,
                  const float* __restrict__ maskB, float* __restrict__ mask_dec,
                  int* __restrict__ pm, float* __restrict__ lps,
                  int* __restrict__ prev_idx, const unsigned* __restrict__ keyv,
                  int t) {
  const int lane = threadIdx.x & 31, w4 = threadIdx.x >> 5;
  const int bm = blockIdx.x * 4 + w4;
  const int b = bm / NS_;
  __shared__ float sq[4][H_];
  __shared__ float sLp[4][SEQ_];
  __shared__ float sP[4][SEQ_];
#pragma unroll
  for (int j = 0; j < 8; ++j)
    sq[w4][j * 32 + lane] = qp[(size_t)bm * H_ + j * 32 + lane];
  __syncthreads();

  float lg[4];
#pragma unroll
  for (int j = 0; j < 4; ++j) {
    const int l = lane + j * 32;
    float s = NEG_;
    if (l < SEQ_ && mask_dec[bm * SEQ_ + l] != 0.0f) {
      const float* kr = kp + (size_t)(l * B_ + b) * H_;
      float d = 0.0f;
      for (int e = 0; e < H_; ++e) d = fmaf(sq[w4][e], kr[e], d);
      s = CLIP_ * tanhf(d * (1.0f / 16.0f));  // 1/sqrt(256)
    }
    lg[j] = s;
  }
  float m = fmaxf(fmaxf(lg[0], lg[1]), fmaxf(lg[2], lg[3]));
  m = wave_max(m);
  float ex[4]; float se = 0.0f;
#pragma unroll
  for (int j = 0; j < 4; ++j) {
    ex[j] = (lg[j] > 0.5f * NEG_) ? __expf(lg[j] - m) : 0.0f;
    se += ex[j];
  }
  se = wave_sum(se);
  const float lse = m + __logf(se);
  const float inv = 1.0f / se;
#pragma unroll
  for (int j = 0; j < 4; ++j) {
    const int l = lane + j * 32;
    if (l < SEQ_) { sLp[w4][l] = lg[j] - lse; sP[w4][l] = ex[j] * inv; }
  }
  __syncthreads();

  if (lane == 0) {
    const unsigned long long seed =
        ((unsigned long long)keyv[0] << 32) ^
        ((unsigned long long)(unsigned)t * 0x9E3779B1ULL) ^
        ((unsigned long long)(unsigned)bm * 0x85EBCA77ULL);
    const float u = u01_hash(seed);
    float c = 0.0f; int chosen = -1; int lastPos = 0;
    for (int l = 0; l < SEQ_; ++l) {
      const float p = sP[w4][l];
      if (p > 0.0f) lastPos = l;
      c += p;
      if (chosen < 0 && u < c) chosen = l;
    }
    if (chosen < 0) chosen = lastPos;
    pm[bm * SEQ_ + t] = chosen;
    lps[bm] += sLp[w4][chosen] * maskB[b * SEQ_ + t];
    prev_idx[bm] = chosen;
    mask_dec[bm * SEQ_ + chosen] = 0.0f;
  }
}

// ---------------------------------------------------------------------------
// Output: pm -> [NS,B,L] int32 (masked -> -1), then lps -> [NS,B] f32.
// ---------------------------------------------------------------------------
__global__ void finalize_kernel(const int* __restrict__ pm, const float* __restrict__ lps,
                                const float* __restrict__ maskB, float* __restrict__ outF) {
  const int idx = blockIdx.x * blockDim.x + threadIdx.x;
  int* outI = (int*)outF;
  if (idx < NS_ * B_ * SEQ_) {
    const int l = idx % SEQ_;
    const int rb = idx / SEQ_;
    const int b = rb % B_;
    const int ns = rb / B_;
    int v = pm[(b * NS_ + ns) * SEQ_ + l];
    if (maskB[b * SEQ_ + l] == 0.0f) v = -1;
    outI[idx] = v;
  }
  if (idx < NS_ * B_) {
    const int b = idx % B_;
    const int ns = idx / B_;
    outF[NS_ * B_ * SEQ_ + idx] = lps[b * NS_ + ns];
  }
}

// f32 -> f16 weight conversion
__global__ void cvt_kernel(const float* __restrict__ in, _Float16* __restrict__ out, int n) {
  const int i = blockIdx.x * blockDim.x + threadIdx.x;
  if (i < n) out[i] = (_Float16)in[i];
}

// ---------------------------------------------------------------------------
// Host orchestration
// ---------------------------------------------------------------------------
struct LayerP {
  const float *b1, *b2, *in_b, *in_w, *n1b, *n1g, *n2b, *n2g, *out_b, *out_w, *w1, *w2;
};
struct ParamsP {
  const float *dec_in_b, *dec_in_w, *dec_out_b, *dec_out_w;
  LayerP layer[NL_];
  const float *proj1_w, *proj_b, *proj_w, *ptr_bk, *ptr_bq, *ptr_wk, *ptr_wq, *v1, *v2;
};

extern "C" void kernel_launch(void* const* d_in, const int* in_sizes, int n_in,
                              void* d_out, int out_size, void* d_ws, size_t ws_size,
                              hipStream_t stream) {
  const float* inp   = (const float*)d_in[0];   // [SEQ,B,IN]
  const float* maskB = (const float*)d_in[1];   // [B,SEQ]

  // --- locate params (JAX pytree leaves, sorted-key recursive order) ---
  const bool packed = (n_in <= 6);              // single flat params buffer at d_in[3]
  const float* pbase = (const float*)d_in[3];
  int li = 0;
  auto leaf = [&](size_t n) -> const float* {
    if (packed) { const float* r = pbase; pbase += n; return r; }
    return (const float*)d_in[3 + (li++)];
  };
  ParamsP P;
  P.dec_in_b  = leaf(3 * H_);
  P.dec_in_w  = leaf((size_t)3 * H_ * H_);
  P.dec_out_b = leaf(H_);
  P.dec_out_w = leaf((size_t)H_ * H_);
  for (int l = 0; l < NL_; ++l) {
    P.layer[l].b1    = leaf(DFF_);
    P.layer[l].b2    = leaf(H_);
    P.layer[l].in_b  = leaf(3 * H_);
    P.layer[l].in_w  = leaf((size_t)3 * H_ * H_);
    P.layer[l].n1b   = leaf(H_);
    P.layer[l].n1g   = leaf(H_);
    P.layer[l].n2b   = leaf(H_);
    P.layer[l].n2g   = leaf(H_);
    P.layer[l].out_b = leaf(H_);
    P.layer[l].out_w = leaf((size_t)H_ * H_);
    P.layer[l].w1    = leaf((size_t)DFF_ * H_);
    P.layer[l].w2    = leaf((size_t)H_ * DFF_);
  }
  P.proj1_w = leaf((size_t)H_ * 3 * H_);
  P.proj_b  = leaf(H_);
  P.proj_w  = leaf((size_t)H_ * IN_);
  P.ptr_bk  = leaf(H_);
  P.ptr_bq  = leaf(H_);
  P.ptr_wk  = leaf((size_t)H_ * H_);
  P.ptr_wq  = leaf((size_t)H_ * H_);
  P.v1      = leaf(H_);
  P.v2      = leaf(H_);
  const unsigned* keyv = packed ? (const unsigned*)d_in[4] : (const unsigned*)d_in[3 + li];

  // --- workspace layout (bump allocator, 256B aligned) ---
  char* wptr = (char*)d_ws;
  auto alloc = [&](size_t bytes) -> void* {
    void* r = (void*)wptr;
    wptr += (bytes + 255) & ~(size_t)255;
    return r;
  };
  // f16 weight copies
  _Float16 *in_w_h[NL_], *out_w_h[NL_], *w1_h[NL_], *w2_h[NL_];
  for (int l = 0; l < NL_; ++l) {
    in_w_h[l]  = (_Float16*)alloc((size_t)3 * H_ * H_ * 2);
    out_w_h[l] = (_Float16*)alloc((size_t)H_ * H_ * 2);
    w1_h[l]    = (_Float16*)alloc((size_t)DFF_ * H_ * 2);
    w2_h[l]    = (_Float16*)alloc((size_t)H_ * DFF_ * 2);
  }
  _Float16* proj1_h   = (_Float16*)alloc((size_t)H_ * 3 * H_ * 2);
  _Float16* dec_in_h  = (_Float16*)alloc((size_t)3 * H_ * H_ * 2);
  _Float16* dec_out_h = (_Float16*)alloc((size_t)H_ * H_ * 2);
  _Float16* ptr_wq_h  = (_Float16*)alloc((size_t)H_ * H_ * 2);
  _Float16* ptr_wk_h  = (_Float16*)alloc((size_t)H_ * H_ * 2);
  // activations
  float*    x      = (float*)alloc((size_t)ROWS_ * H_ * 4);
  _Float16* xh     = (_Float16*)alloc((size_t)ROWS_ * H_ * 2);
  float*    qkv    = (float*)alloc((size_t)ROWS_ * 3 * H_ * 4);
  _Float16* attn_h = (_Float16*)alloc((size_t)ROWS_ * H_ * 2);
  float*    tmp    = (float*)alloc((size_t)ROWS_ * H_ * 4);
  _Float16* ffn_h  = (_Float16*)alloc((size_t)ROWS_ * DFF_ * 2);
  float*    graph  = (float*)alloc((size_t)B_ * H_ * 4);
  float*    Kd     = (float*)alloc((size_t)ROWS_ * H_ * 4);
  float*    Vd     = (float*)alloc((size_t)ROWS_ * H_ * 4);
  float*    kp     = (float*)alloc((size_t)ROWS_ * H_ * 4);
  // decoder state
  _Float16* ctx_h  = (_Float16*)alloc((size_t)BM_ * 3 * H_ * 2);
  _Float16* ctxp_h = (_Float16*)alloc((size_t)BM_ * H_ * 2);
  float*    qdec   = (float*)alloc((size_t)BM_ * H_ * 4);
  _Float16* glh    = (_Float16*)alloc((size_t)BM_ * H_ * 2);
  _Float16* gl_h   = (_Float16*)alloc((size_t)BM_ * H_ * 2);
  float*    qp     = (float*)alloc((size_t)BM_ * H_ * 4);
  float*    mdec   = (float*)alloc((size_t)BM_ * SEQ_ * 4);
  int*      pm     = (int*)alloc((size_t)BM_ * SEQ_ * 4);
  float*    lps    = (float*)alloc((size_t)BM_ * 4);
  int*      pidx   = (int*)alloc((size_t)BM_ * 4);

  auto cvt = [&](const float* src, _Float16* dst, size_t n) {
    cvt_kernel<<<(int)((n + 255) / 256), 256, 0, stream>>>(src, dst, (int)n);
  };
  auto gemm = [&](const _Float16* A, const _Float16* W, const float* bias,
                  float* Cf, _Float16* Ch, int M, int N, int K, int flags) {
    dim3 g(N / 64, M / 64);
    gemm_wmma_kernel<<<g, 128, 0, stream>>>(A, W, bias, Cf, Ch, M, N, K, flags);
  };

  // --- weight conversion (once per call; deterministic) ---
  for (int l = 0; l < NL_; ++l) {
    cvt(P.layer[l].in_w,  in_w_h[l],  (size_t)3 * H_ * H_);
    cvt(P.layer[l].out_w, out_w_h[l], (size_t)H_ * H_);
    cvt(P.layer[l].w1,    w1_h[l],    (size_t)DFF_ * H_);
    cvt(P.layer[l].w2,    w2_h[l],    (size_t)H_ * DFF_);
  }
  cvt(P.proj1_w,  proj1_h,   (size_t)H_ * 3 * H_);
  cvt(P.dec_in_w, dec_in_h,  (size_t)3 * H_ * H_);
  cvt(P.dec_out_w, dec_out_h,(size_t)H_ * H_);
  cvt(P.ptr_wq,   ptr_wq_h,  (size_t)H_ * H_);
  cvt(P.ptr_wk,   ptr_wk_h,  (size_t)H_ * H_);

  // --- encoder ---
  in_proj_kernel<<<(ROWS_ * H_ + 255) / 256, 256, 0, stream>>>(inp, P.proj_w, P.proj_b, x, xh);
  for (int l = 0; l < NL_; ++l) {
    gemm(xh, in_w_h[l], P.layer[l].in_b, qkv, nullptr, ROWS_, 3 * H_, H_, 1);
    enc_attn_kernel<<<(B_ * NH_ * SEQ_) / 4, 128, 0, stream>>>(qkv, maskB, attn_h);
    gemm(attn_h, out_w_h[l], P.layer[l].out_b, tmp, nullptr, ROWS_, H_, H_, 1);
    ln_kernel<<<ROWS_ / 4, 128, 0, stream>>>(x, tmp, P.layer[l].n1g, P.layer[l].n1b, xh, ROWS_);
    gemm(xh, w1_h[l], P.layer[l].b1, nullptr, ffn_h, ROWS_, DFF_, H_, 3);       // +relu
    gemm(ffn_h, w2_h[l], P.layer[l].b2, tmp, nullptr, ROWS_, H_, DFF_, 1);
    ln_kernel<<<ROWS_ / 4, 128, 0, stream>>>(x, tmp, P.layer[l].n2g, P.layer[l].n2b, xh, ROWS_);
  }
  graph_mean_kernel<<<(B_ * H_ + 255) / 256, 256, 0, stream>>>(x, graph);

  // --- decoder constants (hoisted out of the 100-step loop) ---
  gemm(xh, dec_in_h + (size_t)H_ * H_,     P.dec_in_b + H_,     Kd, nullptr, ROWS_, H_, H_, 1);
  gemm(xh, dec_in_h + (size_t)2 * H_ * H_, P.dec_in_b + 2 * H_, Vd, nullptr, ROWS_, H_, H_, 1);
  gemm(xh, ptr_wk_h, P.ptr_bk, kp, nullptr, ROWS_, H_, H_, 1);

  dec_init_kernel<<<(BM_ * SEQ_ + 255) / 256, 256, 0, stream>>>(maskB, mdec, pm, lps, pidx);

  // --- autoregressive decode: 100 sequential steps ---
  for (int t = 0; t < SEQ_; ++t) {
    ctx_build_kernel<<<(BM_ * 3 * H_ + 255) / 256, 256, 0, stream>>>(
        x, graph, P.v1, P.v2, pm, pidx, ctx_h, t);
    gemm(ctx_h, proj1_h, nullptr, nullptr, ctxp_h, BM_, H_, 3 * H_, 0);          // ctx projection
    gemm(ctxp_h, dec_in_h, P.dec_in_b, qdec, nullptr, BM_, H_, H_, 1);           // q projection
    dec_attn_kernel<<<(BM_ * NH_) / 4, 128, 0, stream>>>(qdec, Kd, Vd, mdec, glh);
    gemm(glh, dec_out_h, P.dec_out_b, nullptr, gl_h, BM_, H_, H_, 1);            // out projection
    gemm(gl_h, ptr_wq_h, P.ptr_bq, qp, nullptr, BM_, H_, H_, 1);                 // pointer query
    ptr_sample_kernel<<<BM_ / 4, 128, 0, stream>>>(qp, kp, maskB, mdec, pm, lps, pidx, keyv, t);
  }

  finalize_kernel<<<(NS_ * B_ * SEQ_ + 255) / 256, 256, 0, stream>>>(pm, lps, maskB, (float*)d_out);
}
